// Transformer_91122026152543
// MI455X (gfx1250) — compile-verified, matrix operand order checked by reading, not run on hardware
//
#include <hip/hip_runtime.h>
#include <math.h>

typedef __attribute__((ext_vector_type(16))) __bf16 v16bf;
typedef __attribute__((ext_vector_type(8)))  __bf16 v8bf;
typedef __attribute__((ext_vector_type(8)))  float  v8f;
typedef __attribute__((ext_vector_type(4)))  float  v4f;

#define BATCH 32
#define NTOK  5000
#define DIM   256
#define HEADS 8
#define DH    32
#define MLP   1024
#define DEPTH 4
#define ROWS  (BATCH * NTOK)   // 160000

// ---------------------------------------------------------------------------
// Fragment loaders (CDNA5 wave32 WMMA bf16 16x16x32 layouts, ISA 7.12.2)
// A (16-bit, 16x32): lane L -> row M = L&15; half = L>>4; elements i=0..7 at
// K = kbase + half*8 + i (contiguous), i=8..15 at K = kbase + 16 + half*8 + i-8.
// ---------------------------------------------------------------------------
__device__ __forceinline__ v16bf load_a_frag(const __bf16* base, int rowstride,
                                             int m0, int kbase, int lane) {
  int row = m0 + (lane & 15);
  int ks  = kbase + ((lane >> 4) << 3);
  const __bf16* p = base + row * rowstride + ks;
  union { v16bf v; v8bf h[2]; } u;
  u.h[0] = *(const v8bf*)p;
  u.h[1] = *(const v8bf*)(p + 16);
  return u.v;
}

// B fragments pre-packed at setup: flat = tile*512 + lane*16 + i  (fully coalesced)
__device__ __forceinline__ v16bf load_b_frag(const __bf16* packed, int tile, int lane) {
  const __bf16* p = packed + (size_t)tile * 512 + lane * 16;
  union { v16bf v; v8bf h[2]; } u;
  u.h[0] = *(const v8bf*)p;
  u.h[1] = *(const v8bf*)(p + 8);
  return u.v;
}

// ---------------------------------------------------------------------------
// Pack a row-major fp32 weight [K][Ncols] into WMMA-B fragment-native bf16.
// tile = nt*ktiles + kk ; element (lane,i) holds W[kk*32 + Kf(i,half)][nt*16 + (lane&15)]
// ---------------------------------------------------------------------------
__global__ void pack_b_kernel(const float* __restrict__ w, __bf16* __restrict__ out,
                              int K, int Ncols) {
  int ktiles = K >> 5;
  int total  = K * Ncols;
  int e = blockIdx.x * blockDim.x + threadIdx.x;
  if (e >= total) return;
  int tile = e >> 9;
  int r    = e & 511;
  int lane = r >> 4;
  int i    = r & 15;
  int kk   = tile % ktiles;
  int nt   = tile / ktiles;
  int half = lane >> 4;
  int n    = (nt << 4) + (lane & 15);
  int k    = (kk << 5) + ((i < 8) ? i : (i + 8)) + (half << 3);
  out[e] = (__bf16)w[(size_t)k * Ncols + n];
}

// ---------------------------------------------------------------------------
// q[b,:] = x[b, idx_b, :] @ Wq   (tiny: 32x256 @ 256x256)
// ---------------------------------------------------------------------------
__global__ void q_kernel(const float* __restrict__ x, const int* __restrict__ pidx,
                         const int* __restrict__ hp, const float* __restrict__ Wq,
                         float* __restrict__ q) {
  __shared__ float sel[DIM];
  int b = blockIdx.x, t = threadIdx.x;
  int idx = pidx[2 * b] * hp[0] + pidx[2 * b + 1];
  sel[t] = x[((size_t)b * NTOK + idx) * DIM + t];
  __syncthreads();
  float acc = 0.f;
  for (int k = 0; k < DIM; ++k) acc += sel[k] * Wq[(size_t)k * DIM + t];
  q[b * DIM + t] = acc;
}

// P[b,h,D] = scale * sum_d Wk[D, h*32+d] * q[b, h*32+d]
__global__ void p_kernel(const float* __restrict__ q, const float* __restrict__ Wk,
                         float* __restrict__ P) {
  __shared__ float qr[DIM];
  int b = blockIdx.x, D = threadIdx.x;
  qr[D] = q[b * DIM + D];
  __syncthreads();
  const float scale = 0.17677669529663689f;  // 32^-0.5
  const float* wrow = Wk + (size_t)D * DIM;
  for (int h = 0; h < HEADS; ++h) {
    float a = 0.f;
    for (int d = 0; d < DH; ++d) a += wrow[h * DH + d] * qr[h * DH + d];
    P[(b * HEADS + h) * DIM + D] = a * scale;
  }
}

// dots[b,h,n] = x[b,n,:] . P[b,h,:]   — streaming pass over x, wave-per-token
__global__ void dots_kernel(const float* __restrict__ x, const float* __restrict__ P,
                            float* __restrict__ dots) {
  __shared__ float Pl[HEADS * DIM];
  int b = blockIdx.y, t = threadIdx.x;
  for (int i = t; i < HEADS * DIM; i += 256) Pl[i] = P[b * HEADS * DIM + i];
  __syncthreads();
  int lane = t & 31, wave = t >> 5;
  int n0 = blockIdx.x * 125;
  for (int n = n0 + wave; n < n0 + 125; n += 8) {
    const float* xr = x + ((size_t)b * NTOK + n) * DIM + lane * 8;
    v4f a = *(const v4f*)xr;
    v4f c = *(const v4f*)(xr + 4);
    float acc[HEADS];
#pragma unroll
    for (int h = 0; h < HEADS; ++h) {
      const float* pr = Pl + h * DIM + lane * 8;
      acc[h] = a.x * pr[0] + a.y * pr[1] + a.z * pr[2] + a.w * pr[3] +
               c.x * pr[4] + c.y * pr[5] + c.z * pr[6] + c.w * pr[7];
    }
#pragma unroll
    for (int h = 0; h < HEADS; ++h) {
      float s = acc[h];
#pragma unroll
      for (int off = 16; off; off >>= 1) s += __shfl_xor(s, off, 32);
      if (lane == 0) dots[((size_t)(b * HEADS + h)) * NTOK + n] = s;
    }
  }
}

// softmax over n=5000, one block per (b,h), in place
__global__ void softmax_kernel(float* __restrict__ dots) {
  __shared__ float red[256];
  size_t base = (size_t)blockIdx.x * NTOK;
  int t = threadIdx.x;
  float m = -1e30f;
  for (int n = t; n < NTOK; n += 256) m = fmaxf(m, dots[base + n]);
  red[t] = m; __syncthreads();
  for (int s = 128; s; s >>= 1) { if (t < s) red[t] = fmaxf(red[t], red[t + s]); __syncthreads(); }
  m = red[0]; __syncthreads();
  float sum = 0.f;
  for (int n = t; n < NTOK; n += 256) { float e = expf(dots[base + n] - m); dots[base + n] = e; sum += e; }
  red[t] = sum; __syncthreads();
  for (int s = 128; s; s >>= 1) { if (t < s) red[t] += red[t + s]; __syncthreads(); }
  float inv = 1.f / red[0];
  for (int n = t; n < NTOK; n += 256) dots[base + n] *= inv;
}

__global__ void zero_kernel(float* p, int n) {
  int i = blockIdx.x * blockDim.x + threadIdx.x;
  if (i < n) p[i] = 0.f;
}

// s[b,h,D] = sum_n attn[b,h,n] * x[b,n,D]   — second streaming pass over x
__global__ void wsum_kernel(const float* __restrict__ x, const float* __restrict__ attn,
                            float* __restrict__ s) {
  int b = blockIdx.x, chunk = blockIdx.y, D = threadIdx.x;
  int n0 = chunk * 625;
  float acc[HEADS] = {};
  const float* xb = x + (size_t)b * NTOK * DIM;
  for (int n = n0; n < n0 + 625; ++n) {
    float xv = xb[(size_t)n * DIM + D];
#pragma unroll
    for (int h = 0; h < HEADS; ++h)
      acc[h] += attn[((size_t)(b * HEADS + h)) * NTOK + n] * xv;
  }
  for (int h = 0; h < HEADS; ++h) atomicAdd(&s[(b * HEADS + h) * DIM + D], acc[h]);
}

// attout[b, h*32+d] = sum_D s[b,h,D] * Wv[D, h*32+d]
__global__ void vo_kernel(const float* __restrict__ s, const float* __restrict__ Wv,
                          float* __restrict__ attout) {
  __shared__ float sl[HEADS * DIM];
  int b = blockIdx.x, t = threadIdx.x;
  for (int i = t; i < HEADS * DIM; i += 256) sl[i] = s[b * HEADS * DIM + i];
  __syncthreads();
  int h = t >> 5;
  float acc = 0.f;
  for (int D = 0; D < DIM; ++D) acc += sl[h * DIM + D] * Wv[(size_t)D * DIM + t];
  attout[b * DIM + t] = acc;
}

// x[b, idx_b, :] += attout[b,:] @ Wo + bo
__global__ void oproj_kernel(const float* __restrict__ attout, const float* __restrict__ Wo,
                             const float* __restrict__ bo, const int* __restrict__ pidx,
                             const int* __restrict__ hp, float* __restrict__ x) {
  __shared__ float ol[DIM];
  int b = blockIdx.x, t = threadIdx.x;
  ol[t] = attout[b * DIM + t];
  __syncthreads();
  float acc = bo[t];
  for (int k = 0; k < DIM; ++k) acc += ol[k] * Wo[(size_t)k * DIM + t];
  int idx = pidx[2 * b] * hp[0] + pidx[2 * b + 1];
  x[((size_t)b * NTOK + idx) * DIM + t] += acc;
}

// ---------------------------------------------------------------------------
// Fused FFN: one wave (32 threads) per 32-row tile.
// x -> LN(bf16 in LDS) -> GEMM1(WMMA bf16, +b1, exact GELU, bf16 h in LDS)
//   -> GEMM2(WMMA bf16, +b2) -> residual add into x.   64 KB static LDS
// (LN tile aliases h buffer; GEMM1 A-frags preloaded to registers first).
// ---------------------------------------------------------------------------
__global__ __launch_bounds__(32) void ffn_kernel(
    float* __restrict__ x, const float* __restrict__ lng, const float* __restrict__ lnb,
    const __bf16* __restrict__ w1p, const float* __restrict__ b1,
    const __bf16* __restrict__ w2p, const float* __restrict__ b2) {
  __shared__ __bf16 hbuf[32 * MLP];   // 64 KB; first 16 KB temporarily holds LN tile
  __bf16* xln = hbuf;
  int lane = threadIdx.x;
  size_t row0 = (size_t)blockIdx.x * 32;

  float g[8], bb[8];
#pragma unroll
  for (int j = 0; j < 8; ++j) { g[j] = lng[lane * 8 + j]; bb[j] = lnb[lane * 8 + j]; }

  // LayerNorm 32 rows -> bf16 LDS tile
  for (int r = 0; r < 32; ++r) {
    const float* xr = x + (row0 + r) * DIM + lane * 8;
    float v[8], s = 0.f, s2 = 0.f;
#pragma unroll
    for (int j = 0; j < 8; ++j) { v[j] = xr[j]; s += v[j]; s2 += v[j] * v[j]; }
#pragma unroll
    for (int off = 16; off; off >>= 1) { s += __shfl_xor(s, off, 32); s2 += __shfl_xor(s2, off, 32); }
    float mu = s * (1.f / 256.f);
    float rstd = rsqrtf(s2 * (1.f / 256.f) - mu * mu + 1e-5f);
#pragma unroll
    for (int j = 0; j < 8; ++j)
      xln[r * DIM + lane * 8 + j] = (__bf16)(((v[j] - mu) * rstd) * g[j] + bb[j]);
  }
  __syncthreads();

  // Preload GEMM1 A fragments (K=256 -> 8 steps x 2 row-halves = 128 VGPRs)
  v16bf a1f[8][2];
#pragma unroll
  for (int kk = 0; kk < 8; ++kk) {
    a1f[kk][0] = load_a_frag(xln, DIM, 0,  kk * 32, lane);
    a1f[kk][1] = load_a_frag(xln, DIM, 16, kk * 32, lane);
  }
  __syncthreads();

  int half = lane >> 4, ncol = lane & 15;

  // GEMM1: [32x256] @ [256x1024], bias + exact GELU, bf16 into hbuf
  for (int nt = 0; nt < 64; ++nt) {
    v8f c0 = {}, c1 = {};
#pragma unroll
    for (int kk = 0; kk < 8; ++kk) {
      v16bf bf = load_b_frag(w1p, nt * 8 + kk, lane);
      c0 = __builtin_amdgcn_wmma_f32_16x16x32_bf16(false, a1f[kk][0], false, bf, (short)0, c0, false, false);
      c1 = __builtin_amdgcn_wmma_f32_16x16x32_bf16(false, a1f[kk][1], false, bf, (short)0, c1, false, false);
    }
    int col = nt * 16 + ncol;
    float bias = b1[col];
#pragma unroll
    for (int r = 0; r < 8; ++r) {
      float v0 = c0[r] + bias;
      hbuf[(r + half * 8) * MLP + col]      = (__bf16)(0.5f * v0 * (1.f + erff(v0 * 0.70710678118f)));
      float v1 = c1[r] + bias;
      hbuf[(16 + r + half * 8) * MLP + col] = (__bf16)(0.5f * v1 * (1.f + erff(v1 * 0.70710678118f)));
    }
  }
  __syncthreads();

  // GEMM2: [32x1024] @ [1024x256], bias + residual into x
  for (int nt = 0; nt < 16; ++nt) {
    v8f c0 = {}, c1 = {};
    for (int kk = 0; kk < 32; ++kk) {
      v16bf a0 = load_a_frag(hbuf, MLP, 0,  kk * 32, lane);
      v16bf a1 = load_a_frag(hbuf, MLP, 16, kk * 32, lane);
      v16bf bf = load_b_frag(w2p, nt * 32 + kk, lane);
      c0 = __builtin_amdgcn_wmma_f32_16x16x32_bf16(false, a0, false, bf, (short)0, c0, false, false);
      c1 = __builtin_amdgcn_wmma_f32_16x16x32_bf16(false, a1, false, bf, (short)0, c1, false, false);
    }
    int col = nt * 16 + ncol;
    float bias = b2[col];
#pragma unroll
    for (int r = 0; r < 8; ++r) {
      x[(row0 + r + half * 8) * DIM + col]      += c0[r] + bias;
      x[(row0 + 16 + r + half * 8) * DIM + col] += c1[r] + bias;
    }
  }
}

// Final LayerNorm, in place, wave-per-row
__global__ void finalln_kernel(float* __restrict__ x, const float* __restrict__ g,
                               const float* __restrict__ b) {
  int lane = threadIdx.x & 31, wave = threadIdx.x >> 5;
  size_t row = (size_t)blockIdx.x * 8 + wave;
  float* xr = x + row * DIM + lane * 8;
  float v[8], s = 0.f, s2 = 0.f;
#pragma unroll
  for (int j = 0; j < 8; ++j) { v[j] = xr[j]; s += v[j]; s2 += v[j] * v[j]; }
#pragma unroll
  for (int off = 16; off; off >>= 1) { s += __shfl_xor(s, off, 32); s2 += __shfl_xor(s2, off, 32); }
  float mu = s * (1.f / 256.f);
  float rstd = rsqrtf(s2 * (1.f / 256.f) - mu * mu + 1e-5f);
#pragma unroll
  for (int j = 0; j < 8; ++j)
    xr[j] = (v[j] - mu) * rstd * g[lane * 8 + j] + b[lane * 8 + j];
}

// ---------------------------------------------------------------------------
extern "C" void kernel_launch(void* const* d_in, const int* in_sizes, int n_in,
                              void* d_out, int out_size, void* d_ws, size_t ws_size,
                              hipStream_t stream) {
  const float* x_in = (const float*)d_in[0];
  const int*   pidx = (const int*)d_in[1];
  const float* Wq   = (const float*)d_in[2];
  const float* Wk   = (const float*)d_in[3];
  const float* Wv   = (const float*)d_in[4];
  const float* Wo   = (const float*)d_in[5];
  const float* bo   = (const float*)d_in[6];
  const float* lng  = (const float*)d_in[7];
  const float* lnb  = (const float*)d_in[8];
  const float* W1   = (const float*)d_in[9];
  const float* b1   = (const float*)d_in[10];
  const float* W2   = (const float*)d_in[11];
  const float* b2   = (const float*)d_in[12];
  const float* ng   = (const float*)d_in[13];
  const float* nb   = (const float*)d_in[14];
  const int*   hp   = (const int*)d_in[16];

  float* x = (float*)d_out;
  hipMemcpyAsync(x, x_in, (size_t)ROWS * DIM * sizeof(float),
                 hipMemcpyDeviceToDevice, stream);

  float* q      = (float*)d_ws;
  float* P      = q + BATCH * DIM;
  float* dots   = P + BATCH * HEADS * DIM;
  float* sbuf   = dots + (size_t)BATCH * HEADS * NTOK;
  float* attout = sbuf + BATCH * HEADS * DIM;
  __bf16* w1p   = (__bf16*)(attout + BATCH * DIM);
  __bf16* w2p   = w1p + (size_t)DEPTH * DIM * MLP;

  for (int l = 0; l < DEPTH; ++l) {
    pack_b_kernel<<<(DIM * MLP + 255) / 256, 256, 0, stream>>>(
        W1 + (size_t)l * DIM * MLP, w1p + (size_t)l * DIM * MLP, DIM, MLP);
    pack_b_kernel<<<(MLP * DIM + 255) / 256, 256, 0, stream>>>(
        W2 + (size_t)l * MLP * DIM, w2p + (size_t)l * MLP * DIM, MLP, DIM);
  }

  for (int l = 0; l < DEPTH; ++l) {
    q_kernel<<<BATCH, 256, 0, stream>>>(x, pidx, hp, Wq + (size_t)l * DIM * DIM, q);
    p_kernel<<<BATCH, 256, 0, stream>>>(q, Wk + (size_t)l * DIM * DIM, P);
    dots_kernel<<<dim3(40, BATCH), 256, 0, stream>>>(x, P, dots);
    softmax_kernel<<<BATCH * HEADS, 256, 0, stream>>>(dots);
    zero_kernel<<<(BATCH * HEADS * DIM + 255) / 256, 256, 0, stream>>>(sbuf, BATCH * HEADS * DIM);
    wsum_kernel<<<dim3(BATCH, 8), 256, 0, stream>>>(x, dots, sbuf);
    vo_kernel<<<BATCH, 256, 0, stream>>>(sbuf, Wv + (size_t)l * DIM * DIM, attout);
    oproj_kernel<<<BATCH, 256, 0, stream>>>(attout, Wo + (size_t)l * DIM * DIM,
                                            bo + l * DIM, pidx, hp, x);
    ffn_kernel<<<ROWS / 32, 32, 0, stream>>>(x, lng + l * DIM, lnb + l * DIM,
                                             w1p + (size_t)l * DIM * MLP, b1 + l * MLP,
                                             w2p + (size_t)l * MLP * DIM, b2 + l * DIM);
  }

  finalln_kernel<<<ROWS / 8, 256, 0, stream>>>(x, ng, nb);
}
